// DGNN_SGS_Conv_6914897347185
// MI455X (gfx1250) — matrix operand order repaired and checked by reading, not compiled, not run on hardware
//
#include <hip/hip_runtime.h>
#include <hip/hip_bf16.h>

#define N_NODES 50000
#define E_EDGES 400000
#define IN_CH   512
#define H_CH    512
#define OUT_CH  256
#define CHUNK   8
#define LN_EPS  1e-5f

typedef __attribute__((ext_vector_type(16))) __bf16 v16bf;
typedef __attribute__((ext_vector_type(8)))  float  v8f;

union FragAB { v16bf v; unsigned int u[8]; };

__device__ __forceinline__ float waveAllSum(float v) {
#pragma unroll
    for (int o = 16; o > 0; o >>= 1) v += __shfl_xor(v, o, 32);
    return v;
}

__device__ __forceinline__ void loadFrag(FragAB& f, const unsigned int* __restrict__ p,
                                         int k0, const int* kk) {
#pragma unroll
    for (int v = 0; v < 8; ++v) f.u[v] = p[(k0 + kk[v]) >> 1];
}

// ---------------------------------------------------------------------------
// bf16 WMMA GEMM: C[M x Nout] = A[M x K] * Bt[Nout x K]^T + bias, optional ReLU
// One wave computes a 16(M) x 64(N) strip (A fragment reused across 4 WMMAs).
// K is a compile-time constant so all fragment addresses become immediate
// offsets. Explicit double buffering: all loads for chunk k+32 are issued
// before the 4 WMMAs of chunk k, so waits become partial instead of full
// drains (breaks the load->wmma serialization seen in round 2).
// ---------------------------------------------------------------------------
template <int K>
__global__ __launch_bounds__(256) void gemm_bf16_wmma(
    const __hip_bfloat16* __restrict__ A,
    const __hip_bfloat16* __restrict__ Bt,
    const float* __restrict__ bias,
    float* __restrict__ C,
    int M, int Nout, int doRelu)
{
    const int strips = Nout >> 6;                       // 64-wide strips
    const int wid = (int)((blockIdx.x * blockDim.x + threadIdx.x) >> 5);
    const int totalW = (M >> 4) * strips;
    if (wid >= totalW) return;
    const int tileM = wid / strips;
    const int strip = wid % strips;
    const int lane = threadIdx.x & 31;
    const int g = lane >> 4;      // half-wave
    const int r = lane & 15;

    // A fragment (16x32 bf16): VGPR v<4 -> K = g*8+2v ; v>=4 -> K = 16+g*8+2(v-4)
    // B fragment (32x16 bf16): VGPR v<4 -> K = g*8+2v ; v>=4 -> K = 16+g*8+2(v-4) (lane = N)
    int kA[8], kB[8];
#pragma unroll
    for (int v = 0; v < 8; ++v) {
        kA[v] = (v < 4) ? (g * 8 + 2 * v) : (16 + g * 8 + 2 * (v - 4));
        kB[v] = kA[v];
    }

    const unsigned int* pA = (const unsigned int*)(A + (size_t)(tileM * 16 + r) * K);
    const unsigned int* pB0 = (const unsigned int*)(Bt + (size_t)(strip * 64 + r) * K);
    const unsigned int* pB1 = pB0 + (16 * K) / 2;
    const unsigned int* pB2 = pB0 + (32 * K) / 2;
    const unsigned int* pB3 = pB0 + (48 * K) / 2;

    FragAB aC, b0C, b1C, b2C, b3C;      // current chunk
    FragAB aN, b0N, b1N, b2N, b3N;      // next chunk (prefetch)
    loadFrag(aC, pA, 0, kA);
    loadFrag(b0C, pB0, 0, kB);
    loadFrag(b1C, pB1, 0, kB);
    loadFrag(b2C, pB2, 0, kB);
    loadFrag(b3C, pB3, 0, kB);

    v8f acc0 = {}, acc1 = {}, acc2 = {}, acc3 = {};
#pragma unroll 2
    for (int k0 = 0; k0 < K; k0 += 32) {
        const int kn = (k0 + 32 < K) ? (k0 + 32) : 0;   // harmless dummy prefetch on last
        loadFrag(aN, pA, kn, kA);
        loadFrag(b0N, pB0, kn, kB);
        loadFrag(b1N, pB1, kn, kB);
        loadFrag(b2N, pB2, kn, kB);
        loadFrag(b3N, pB3, kn, kB);

        acc0 = __builtin_amdgcn_wmma_f32_16x16x32_bf16(false, aC.v, false, b0C.v,
                                                       (short)0, acc0, false, false);
        acc1 = __builtin_amdgcn_wmma_f32_16x16x32_bf16(false, aC.v, false, b1C.v,
                                                       (short)0, acc1, false, false);
        acc2 = __builtin_amdgcn_wmma_f32_16x16x32_bf16(false, aC.v, false, b2C.v,
                                                       (short)0, acc2, false, false);
        acc3 = __builtin_amdgcn_wmma_f32_16x16x32_bf16(false, aC.v, false, b3C.v,
                                                       (short)0, acc3, false, false);

        aC = aN; b0C = b0N; b1C = b1N; b2C = b2N; b3C = b3N;   // renamed away
    }

    // C layout: lane half g, VGPR i -> row g*8+i ; lane r -> col r
    v8f accs[4] = {acc0, acc1, acc2, acc3};
#pragma unroll
    for (int t = 0; t < 4; ++t) {
        const int n = strip * 64 + t * 16 + r;
        const float bv = bias[n];
#pragma unroll
        for (int i = 0; i < 8; ++i) {
            int m = tileM * 16 + g * 8 + i;
            float val = accs[t][i] + bv;
            if (doRelu) val = fmaxf(val, 0.0f);
            C[(size_t)m * Nout + n] = val;
        }
    }
}

// ---------------------------------------------------------------------------
// Helper / elementwise kernels
// ---------------------------------------------------------------------------
__global__ void f32_to_bf16(const float* __restrict__ a,
                            __hip_bfloat16* __restrict__ b, size_t n)
{
    size_t i = (size_t)blockIdx.x * blockDim.x + threadIdx.x;
    size_t stride = (size_t)gridDim.x * blockDim.x;
    for (; i < n; i += stride) b[i] = __float2bfloat16(a[i]);
}

// W: [K x Nout] row-major (fp32) -> Wt: [Nout x K] row-major (bf16)
__global__ void transpose_to_bf16(const float* __restrict__ W,
                                  __hip_bfloat16* __restrict__ Wt,
                                  int K, int Nout)
{
    size_t total = (size_t)K * Nout;
    size_t i = (size_t)blockIdx.x * blockDim.x + threadIdx.x;
    size_t stride = (size_t)gridDim.x * blockDim.x;
    for (; i < total; i += stride) {
        int n = (int)(i / K);
        int k = (int)(i % K);
        Wt[i] = __float2bfloat16(W[(size_t)k * Nout + n]);
    }
}

__global__ void zero_f32(float* __restrict__ p, size_t n)
{
    size_t i = (size_t)blockIdx.x * blockDim.x + threadIdx.x;
    size_t stride = (size_t)gridDim.x * blockDim.x;
    for (; i < n; i += stride) p[i] = 0.0f;
}

// One wave per row: LayerNorm over H_CH=512. Lane l owns float4 groups
// l + 32*j (j=0..3) -> fully coalesced b128 traffic.
__global__ __launch_bounds__(256) void layernorm_rows(
    const float* __restrict__ X, const float* __restrict__ gamma,
    const float* __restrict__ beta, float* __restrict__ Y, int rows)
{
    int row = (int)((blockIdx.x * blockDim.x + threadIdx.x) >> 5);
    if (row >= rows) return;
    int lane = threadIdx.x & 31;
    const float4* xr = (const float4*)(X + (size_t)row * H_CH);
    float4 v[4];
    float s = 0.0f;
#pragma unroll
    for (int j = 0; j < 4; ++j) {
        v[j] = xr[lane + 32 * j];
        s += v[j].x + v[j].y + v[j].z + v[j].w;
    }
    s = waveAllSum(s) * (1.0f / H_CH);
    float var = 0.0f;
#pragma unroll
    for (int j = 0; j < 4; ++j) {
        float dx = v[j].x - s, dy = v[j].y - s, dz = v[j].z - s, dw = v[j].w - s;
        var += dx * dx + dy * dy + dz * dz + dw * dw;
    }
    var = waveAllSum(var) * (1.0f / H_CH);
    float inv = rsqrtf(var + LN_EPS);
    const float4* gg = (const float4*)gamma;
    const float4* bb = (const float4*)beta;
    float4* yr = (float4*)(Y + (size_t)row * H_CH);
#pragma unroll
    for (int j = 0; j < 4; ++j) {
        int q = lane + 32 * j;
        float4 gq = gg[q], bq = bb[q], o;
        o.x = gq.x * (v[j].x - s) * inv + bq.x;
        o.y = gq.y * (v[j].y - s) * inv + bq.y;
        o.z = gq.z * (v[j].z - s) * inv + bq.z;
        o.w = gq.w * (v[j].w - s) * inv + bq.w;
        yr[q] = o;
    }
}

// Edge scatter: each work item moves 4 floats of one edge (gather + atomic add).
// The item handling float4 group 0 (q == 0) also bumps the neighbor count (fused).
__global__ void edge_scatter(const int* __restrict__ src, const int* __restrict__ dst,
                             const float* __restrict__ X, float* __restrict__ S,
                             float* __restrict__ cnt)
{
    const size_t total = (size_t)E_EDGES * (H_CH / 4);
    size_t i = (size_t)blockIdx.x * blockDim.x + threadIdx.x;
    size_t stride = (size_t)gridDim.x * blockDim.x;
    for (; i < total; i += stride) {
        size_t e = i >> 7;              // H_CH/4 == 128
        int q = (int)(i & 127) * 4;
        int sn = src[e], dn = dst[e];
        if (sn == dn) continue;         // remove_self_loops
        const float4 val = *(const float4*)(X + (size_t)sn * H_CH + q);
        float* sp = S + (size_t)dn * H_CH + q;
        atomicAdd(sp + 0, val.x);
        atomicAdd(sp + 1, val.y);
        atomicAdd(sp + 2, val.z);
        atomicAdd(sp + 3, val.w);
        if (q == 0) atomicAdd(&cnt[dn], 1.0f);
    }
}

// One wave per node: mean, gating GEMV ([h,m] @ tmW -> 8), blend, LayerNorm.
// float4 lane layout: lane l owns elements 4*(l+32*j)..+3, j=0..3.
__global__ __launch_bounds__(256) void conv_post(
    const float* __restrict__ X, const float* __restrict__ S,
    const float* __restrict__ cnt,
    const float* __restrict__ tmW, const float* __restrict__ tmB,
    const float* __restrict__ lnG, const float* __restrict__ lnB,
    float* __restrict__ Y)
{
    int node = (int)((blockIdx.x * blockDim.x + threadIdx.x) >> 5);
    if (node >= N_NODES) return;
    int lane = threadIdx.x & 31;
    const float4* xr = (const float4*)(X + (size_t)node * H_CH);
    const float4* sr = (const float4*)(S + (size_t)node * H_CH);
    float inv_c = 1.0f / (cnt[node] + 1.0f);

    float4 h4[4], m4[4];
    float acc[CHUNK];
#pragma unroll
    for (int c = 0; c < CHUNK; ++c) acc[c] = 0.0f;

#pragma unroll
    for (int j = 0; j < 4; ++j) {
        int q = lane + 32 * j;          // float4 group index; k0 = 4*q
        h4[j] = xr[q];
        float4 s4 = sr[q];
        m4[j].x = (s4.x + h4[j].x) * inv_c;
        m4[j].y = (s4.y + h4[j].y) * inv_c;
        m4[j].z = (s4.z + h4[j].z) * inv_c;
        m4[j].w = (s4.w + h4[j].w) * inv_c;
        const float* hp = &h4[j].x;
        const float* mp = &m4[j].x;
#pragma unroll
        for (int c4 = 0; c4 < 4; ++c4) {
            int k = 4 * q + c4;
            const float4* w0 = (const float4*)(tmW + (size_t)k * CHUNK);
            const float4* w1 = (const float4*)(tmW + (size_t)(H_CH + k) * CHUNK);
            float hk = hp[c4], mk = mp[c4];
            float4 wa = w0[0], wb = w0[1], wc = w1[0], wd = w1[1];
            acc[0] = fmaf(hk, wa.x, fmaf(mk, wc.x, acc[0]));
            acc[1] = fmaf(hk, wa.y, fmaf(mk, wc.y, acc[1]));
            acc[2] = fmaf(hk, wa.z, fmaf(mk, wc.z, acc[2]));
            acc[3] = fmaf(hk, wa.w, fmaf(mk, wc.w, acc[3]));
            acc[4] = fmaf(hk, wb.x, fmaf(mk, wd.x, acc[4]));
            acc[5] = fmaf(hk, wb.y, fmaf(mk, wd.y, acc[5]));
            acc[6] = fmaf(hk, wb.z, fmaf(mk, wd.z, acc[6]));
            acc[7] = fmaf(hk, wb.w, fmaf(mk, wd.w, acc[7]));
        }
    }
    float gate[CHUNK];
#pragma unroll
    for (int c = 0; c < CHUNK; ++c) {
        float t = waveAllSum(acc[c]) + tmB[c];
        gate[c] = 1.0f / (1.0f + __expf(-t));
    }

    // blend (channel uniform within an aligned float4; REP=64) + LayerNorm
    float4 o[4];
    float s1 = 0.0f;
#pragma unroll
    for (int j = 0; j < 4; ++j) {
        int q = lane + 32 * j;
        float tg = gate[(4 * q) >> 6];
        o[j].x = h4[j].x * tg + m4[j].x * (1.0f - tg);
        o[j].y = h4[j].y * tg + m4[j].y * (1.0f - tg);
        o[j].z = h4[j].z * tg + m4[j].z * (1.0f - tg);
        o[j].w = h4[j].w * tg + m4[j].w * (1.0f - tg);
        s1 += o[j].x + o[j].y + o[j].z + o[j].w;
    }
    s1 = waveAllSum(s1) * (1.0f / H_CH);
    float var = 0.0f;
#pragma unroll
    for (int j = 0; j < 4; ++j) {
        float dx = o[j].x - s1, dy = o[j].y - s1, dz = o[j].z - s1, dw = o[j].w - s1;
        var += dx * dx + dy * dy + dz * dz + dw * dw;
    }
    var = waveAllSum(var) * (1.0f / H_CH);
    float inv = rsqrtf(var + LN_EPS);
    const float4* gg = (const float4*)lnG;
    const float4* bb = (const float4*)lnB;
    float4* yr = (float4*)(Y + (size_t)node * H_CH);
#pragma unroll
    for (int j = 0; j < 4; ++j) {
        int q = lane + 32 * j;
        float4 gq = gg[q], bq = bb[q], w;
        w.x = gq.x * (o[j].x - s1) * inv + bq.x;
        w.y = gq.y * (o[j].y - s1) * inv + bq.y;
        w.z = gq.z * (o[j].z - s1) * inv + bq.z;
        w.w = gq.w * (o[j].w - s1) * inv + bq.w;
        yr[q] = w;
    }
}

// ---------------------------------------------------------------------------
extern "C" void kernel_launch(void* const* d_in, const int* in_sizes, int n_in,
                              void* d_out, int out_size, void* d_ws, size_t ws_size,
                              hipStream_t stream)
{
    (void)in_sizes; (void)n_in; (void)out_size; (void)ws_size;
    const float* x       = (const float*)d_in[0];
    const int*   eidx    = (const int*)d_in[1];
    const float* W_in    = (const float*)d_in[2];
    const float* b_in    = (const float*)d_in[3];
    const float* ln_in_g = (const float*)d_in[4];
    const float* ln_in_b = (const float*)d_in[5];
    const float* tm_W    = (const float*)d_in[6];
    const float* tm_b    = (const float*)d_in[7];
    const float* ln1_g   = (const float*)d_in[8];
    const float* ln1_b   = (const float*)d_in[9];
    const float* ln2_g   = (const float*)d_in[10];
    const float* ln2_b   = (const float*)d_in[11];
    const float* W_out   = (const float*)d_in[12];
    const float* b_out   = (const float*)d_in[13];
    float* out = (float*)d_out;
    const int* src = eidx;
    const int* dst = eidx + E_EDGES;

    char* ws = (char*)d_ws;
    size_t off = 0;
    auto take = [&](size_t bytes) {
        char* p = ws + off;
        off = (off + bytes + 255) & ~(size_t)255;
        return p;
    };
    __hip_bfloat16* ab    = (__hip_bfloat16*)take((size_t)N_NODES * H_CH * 2);
    __hip_bfloat16* wtin  = (__hip_bfloat16*)take((size_t)H_CH * IN_CH * 2);
    __hip_bfloat16* wtout = (__hip_bfloat16*)take((size_t)OUT_CH * H_CH * 2);
    float* hA  = (float*)take((size_t)N_NODES * H_CH * 4);
    float* hB  = (float*)take((size_t)N_NODES * H_CH * 4);
    float* sb  = (float*)take((size_t)N_NODES * H_CH * 4);
    float* cnt = (float*)take((size_t)N_NODES * 4);

    // ---- input projection: h = LN(ReLU(x @ W_in + b_in)) ----
    f32_to_bf16<<<2048, 256, 0, stream>>>(x, ab, (size_t)N_NODES * IN_CH);
    transpose_to_bf16<<<1024, 256, 0, stream>>>(W_in, wtin, IN_CH, H_CH);
    transpose_to_bf16<<<512, 256, 0, stream>>>(W_out, wtout, H_CH, OUT_CH);

    {   // 16x64 strips: waves = (M/16)*(Nout/64) = 3125*8 = 25000 -> 3125 blocks
        int waves = (N_NODES / 16) * (H_CH / 64);
        gemm_bf16_wmma<IN_CH><<<(waves + 7) / 8, 256, 0, stream>>>(
            ab, wtin, b_in, hB, N_NODES, H_CH, 1);
    }
    layernorm_rows<<<N_NODES / 8, 256, 0, stream>>>(hB, ln_in_g, ln_in_b, hA, N_NODES);

    // ---- conv layer 1: hA -> hB ----
    zero_f32<<<2048, 256, 0, stream>>>(sb, (size_t)N_NODES * H_CH);
    zero_f32<<<256, 256, 0, stream>>>(cnt, (size_t)N_NODES);
    edge_scatter<<<8192, 256, 0, stream>>>(src, dst, hA, sb, cnt);
    conv_post<<<N_NODES / 8, 256, 0, stream>>>(hA, sb, cnt, tm_W, tm_b,
                                               ln1_g, ln1_b, hB);

    // ---- conv layer 2: hB -> hA ----
    zero_f32<<<2048, 256, 0, stream>>>(sb, (size_t)N_NODES * H_CH);
    zero_f32<<<256, 256, 0, stream>>>(cnt, (size_t)N_NODES);
    edge_scatter<<<8192, 256, 0, stream>>>(src, dst, hB, sb, cnt);
    conv_post<<<N_NODES / 8, 256, 0, stream>>>(hB, sb, cnt, tm_W, tm_b,
                                               ln2_g, ln2_b, hA);

    // ---- output projection: out = h @ W_out + b_out ----
    f32_to_bf16<<<2048, 256, 0, stream>>>(hA, ab, (size_t)N_NODES * H_CH);
    {   // waves = 3125*4 = 12500 -> 1563 blocks (last wave-group guarded)
        int waves = (N_NODES / 16) * (OUT_CH / 64);
        gemm_bf16_wmma<H_CH><<<(waves + 7) / 8, 256, 0, stream>>>(
            ab, wtout, b_out, out, N_NODES, OUT_CH, 0);
    }
}
